// Seq2Seq_26139170963781
// MI455X (gfx1250) — matrix operand order repaired and checked by reading, not compiled, not run on hardware
//
#include <hip/hip_runtime.h>
#include <hip/hip_bf16.h>

typedef __attribute__((ext_vector_type(16))) _Float16 v16h;
typedef __attribute__((ext_vector_type(8)))  float    v8f;

#define HREAL   60
#define GW      192          // padded 3*H (3 x 64)
#define BATCH   65536
#define TOBS    8
#define TPRED   12
#define WAVES   4
#define NTHREADS (WAVES*32)

// f16-unit offsets inside dynamic LDS
#define OFF_E0_IH 0          // 32x192
#define OFF_E0_HH 6144       // 64x192
#define OFF_E1_IH 18432      // 64x192
#define OFF_E1_HH 30720      // 64x192
#define OFF_D0_IH 43008      // 32x192
#define OFF_D0_HH 49152      // 64x192
#define OFF_D1_IH 61440      // 64x192
#define OFF_D1_HH 73728      // 64x192
#define OFF_LIN   86016      // 64x16
#define OFF_W_END 87040
#define NBIASF    (8*192+16)                 // floats
#define OFF_TILES (OFF_W_END + NBIASF*2)     // f16 units
#define TILE_F16  2560                       // h0(16x64)+h1(16x64)+x(16x32)
#define SMEM_BYTES ((size_t)(OFF_TILES + WAVES*TILE_F16)*2)

__device__ __forceinline__ v8f splat8(float v){
  v8f r;
#pragma unroll
  for (int i = 0; i < 8; ++i) r[i] = v;
  return r;
}

// Fast activations: v_exp_f32 + v_rcp_f32, no IEEE divide sequences.
__device__ __forceinline__ float rcp_(float x){
  return __builtin_amdgcn_rcpf(x);
}
__device__ __forceinline__ float sigmoid_(float x){
  return rcp_(1.0f + __expf(-x));           // exp(-inf)=0 -> 1 ; exp(inf)=inf -> rcp=0
}
__device__ __forceinline__ float tanh_(float x){
  float e = __expf(2.0f * x);               // inf -> rcp=0 -> +1 ; 0 -> 1-2*rcp(1) = -1
  return 1.0f - 2.0f * rcp_(e + 1.0f);
}

__device__ __forceinline__ v8f wmma16(v16h a, v16h b, v8f c){
  // D = A(16x32 f16) x B(32x16 f16) + C(16x16 f32)
  return __builtin_amdgcn_wmma_f32_16x16x32_f16(
      false, a, false, b, (short)0, c, false, false);
}

// A fragment: lane<16 -> row=lane, K=s*32..+15 ; lane>=16 -> row=lane-16, K=s*32+16..+31
__device__ __forceinline__ v16h loadA(const _Float16* base, int rowF16, int s, int lane){
  const int row = lane & 15;
  const int kh  = lane >> 4;
  return *(const v16h*)(base + row * rowF16 + s * 32 + kh * 16);
}
// B fragment from W^T (K x 192): lane holds column N = tile*16+(lane&15), K = s*32+(lane>>4)*16..+15
__device__ __forceinline__ v16h loadB(const _Float16* w, int tile, int s, int lane){
  v16h r;
#pragma unroll
  for (int e = 0; e < 16; ++e) r[e] = w[(s * 32 + (lane >> 4) * 16 + e) * GW + tile * 16 + (lane & 15)];
  return r;
}
// B fragment from lin W^T (64 x 16)
__device__ __forceinline__ v16h loadBlin(const _Float16* w, int s, int lane){
  v16h r;
#pragma unroll
  for (int e = 0; e < 16; ++e) r[e] = w[(s * 32 + (lane >> 4) * 16 + e) * 16 + (lane & 15)];
  return r;
}

// Stage W (3H x in) row-major f32 -> padded W^T (Kp x 192) f16, gate-padded to 64
__device__ __forceinline__ void stage_w(_Float16* dst, const float* W, int inReal, int tid){
  const int Kp = (inReal <= 2) ? 32 : 64;
  const int total = Kp * GW;
  for (int i = tid; i < total; i += NTHREADS){
    int k = i / GW, g = i - k * GW;
    int gate = g >> 6, u = g & 63;
    float v = 0.0f;
    if (k < inReal && u < HREAL) v = W[(gate * HREAL + u) * inReal + k];
    dst[i] = (_Float16)v;
  }
}
__device__ __forceinline__ void stage_b(float* dst, const float* b, int tid){
  for (int i = tid; i < GW; i += NTHREADS){
    int gate = i >> 6, u = i & 63;
    dst[i] = (u < HREAL) ? b[gate * HREAL + u] : 0.0f;
  }
}

// One GRU cell for a 16-row batch tile. x tile: 16 x xRS f16 (zero padded),
// h tile: 16 x 64 f16 (zero padded). Writes h' back to ht and to hC (C-layout).
template <int XKS>
__device__ __forceinline__ void gru_cell(int lane,
    const _Float16* xt, int xRS,
    _Float16* ht,
    const _Float16* Wih, const _Float16* Whh,
    const float* bih, const float* bhh,
    float hC[4][8])
{
  v16h ax[XKS];
#pragma unroll
  for (int s = 0; s < XKS; ++s) ax[s] = loadA(xt, xRS, s, lane);
  v16h ah[2];
  ah[0] = loadA(ht, 64, 0, lane);
  ah[1] = loadA(ht, 64, 1, lane);

  const int c  = lane & 15;
  const int mb = (lane >> 4) * 8;

  v8f R[4], Z[4], N[4];
#pragma unroll
  for (int t = 0; t < 4; ++t){
    // r gate: tiles 0..3
    v8f acc = splat8(bih[t * 16 + c] + bhh[t * 16 + c]);
#pragma unroll
    for (int s = 0; s < XKS; ++s) acc = wmma16(ax[s], loadB(Wih, t, s, lane), acc);
#pragma unroll
    for (int s = 0; s < 2;   ++s) acc = wmma16(ah[s], loadB(Whh, t, s, lane), acc);
#pragma unroll
    for (int i = 0; i < 8; ++i) R[t][i] = sigmoid_(acc[i]);

    // z gate: tiles 4..7
    acc = splat8(bih[64 + t * 16 + c] + bhh[64 + t * 16 + c]);
#pragma unroll
    for (int s = 0; s < XKS; ++s) acc = wmma16(ax[s], loadB(Wih, t + 4, s, lane), acc);
#pragma unroll
    for (int s = 0; s < 2;   ++s) acc = wmma16(ah[s], loadB(Whh, t + 4, s, lane), acc);
#pragma unroll
    for (int i = 0; i < 8; ++i) Z[t][i] = sigmoid_(acc[i]);

    // n gate: tiles 8..11, gx and gh kept separate (need r * gh_n)
    v8f gx = splat8(bih[128 + t * 16 + c]);
#pragma unroll
    for (int s = 0; s < XKS; ++s) gx = wmma16(ax[s], loadB(Wih, t + 8, s, lane), gx);
    v8f gh = splat8(bhh[128 + t * 16 + c]);
#pragma unroll
    for (int s = 0; s < 2;   ++s) gh = wmma16(ah[s], loadB(Whh, t + 8, s, lane), gh);
#pragma unroll
    for (int i = 0; i < 8; ++i) N[t][i] = tanh_(gx[i] + R[t][i] * gh[i]);
  }

  // h' = (1-z)*n + z*h  (read old h before overwriting; per-lane column is private)
#pragma unroll
  for (int t = 0; t < 4; ++t){
#pragma unroll
    for (int m = 0; m < 8; ++m){
      float hold = (float)ht[(mb + m) * 64 + t * 16 + c];
      float hn = (1.0f - Z[t][m]) * N[t][m] + Z[t][m] * hold;
      hC[t][m] = hn;
      ht[(mb + m) * 64 + t * 16 + c] = (_Float16)hn;
    }
  }
}

__global__ void __launch_bounds__(NTHREADS)
seq2seq_gru_wmma(
    const float* __restrict__ in_data, const float* __restrict__ last_loc,
    const float* e0_Wih, const float* e0_Whh, const float* e0_bih, const float* e0_bhh,
    const float* e1_Wih, const float* e1_Whh, const float* e1_bih, const float* e1_bhh,
    const float* d0_Wih, const float* d0_Whh, const float* d0_bih, const float* d0_bhh,
    const float* d1_Wih, const float* d1_Whh, const float* d1_bih, const float* d1_bhh,
    const float* lin_w, const float* lin_b,
    float* __restrict__ out)
{
  extern __shared__ _Float16 smem[];
  const int tid = threadIdx.x;

  // ---- stage all weights into LDS (f16, padded, transposed) ----
  stage_w(smem + OFF_E0_IH, e0_Wih, 2,  tid);
  stage_w(smem + OFF_E0_HH, e0_Whh, 60, tid);
  stage_w(smem + OFF_E1_IH, e1_Wih, 60, tid);
  stage_w(smem + OFF_E1_HH, e1_Whh, 60, tid);
  stage_w(smem + OFF_D0_IH, d0_Wih, 2,  tid);
  stage_w(smem + OFF_D0_HH, d0_Whh, 60, tid);
  stage_w(smem + OFF_D1_IH, d1_Wih, 60, tid);
  stage_w(smem + OFF_D1_HH, d1_Whh, 60, tid);
  for (int i = tid; i < 64 * 16; i += NTHREADS){
    int k = i >> 4, cc = i & 15;
    smem[OFF_LIN + i] = (_Float16)((k < HREAL && cc < 2) ? lin_w[cc * HREAL + k] : 0.0f);
  }
  float* bias = (float*)(smem + OFF_W_END);
  stage_b(bias + 0 * 192, e0_bih, tid);
  stage_b(bias + 1 * 192, e0_bhh, tid);
  stage_b(bias + 2 * 192, e1_bih, tid);
  stage_b(bias + 3 * 192, e1_bhh, tid);
  stage_b(bias + 4 * 192, d0_bih, tid);
  stage_b(bias + 5 * 192, d0_bhh, tid);
  stage_b(bias + 6 * 192, d1_bih, tid);
  stage_b(bias + 7 * 192, d1_bhh, tid);
  for (int i = tid; i < 16; i += NTHREADS) bias[8 * 192 + i] = (i < 2) ? lin_b[i] : 0.0f;
  // zero activation tiles (h0 = h1 = 0, x pad = 0)
  for (int i = tid; i < WAVES * TILE_F16; i += NTHREADS) smem[OFF_TILES + i] = (_Float16)0.0f;
  __syncthreads();

  const int wave = tid >> 5, lane = tid & 31;
  const long b0 = ((long)blockIdx.x * WAVES + wave) * 16;
  _Float16* h0 = smem + OFF_TILES + wave * TILE_F16;
  _Float16* h1 = h0 + 1024;
  _Float16* xt = h1 + 1024;     // 16 x 32

  const float* bE0ih = bias + 0 * 192, *bE0hh = bias + 1 * 192;
  const float* bE1ih = bias + 2 * 192, *bE1hh = bias + 3 * 192;
  const float* bD0ih = bias + 4 * 192, *bD0hh = bias + 5 * 192;
  const float* bD1ih = bias + 6 * 192, *bD1hh = bias + 7 * 192;
  const float* blin  = bias + 8 * 192;

  float hC0[4][8], hC1[4][8];
  const int c  = lane & 15;
  const int mb = (lane >> 4) * 8;

  // ---- encoder ----
  for (int t = 0; t < TOBS; ++t){
    if (lane < 16){
      const float* xp = in_data + (b0 + lane) * (TOBS * 2) + t * 2;
      xt[lane * 32 + 0] = (_Float16)xp[0];
      xt[lane * 32 + 1] = (_Float16)xp[1];
    }
    gru_cell<1>(lane, xt, 32, h0, smem + OFF_E0_IH, smem + OFF_E0_HH, bE0ih, bE0hh, hC0);
    gru_cell<2>(lane, h0, 64, h1, smem + OFF_E1_IH, smem + OFF_E1_HH, bE1ih, bE1hh, hC1);
  }

  // ---- decoder: x0 = last_location ----
  if (lane < 16){
    const float* xp = last_loc + (b0 + lane) * 2;
    xt[lane * 32 + 0] = (_Float16)xp[0];
    xt[lane * 32 + 1] = (_Float16)xp[1];
  }
  float* hidO = out + (size_t)BATCH * TPRED * 2;

  for (int t = 0; t < TPRED; ++t){
    gru_cell<1>(lane, xt, 32, h0, smem + OFF_D0_IH, smem + OFF_D0_HH, bD0ih, bD0hh, hC0);
    gru_cell<2>(lane, h0, 64, h1, smem + OFF_D1_IH, smem + OFF_D1_HH, bD1ih, bD1hh, hC1);

    // hidden_outputs: (B, TPRED, 120) = [h_l0 | h_l1]
#pragma unroll
    for (int tt = 0; tt < 4; ++tt){
      const int col = tt * 16 + c;
      if (col < HREAL){
#pragma unroll
        for (int m = 0; m < 8; ++m){
          const size_t r = (size_t)(b0 + mb + m);
          hidO[r * (TPRED * 120) + t * 120 + col]        = hC0[tt][m];
          hidO[r * (TPRED * 120) + t * 120 + 60 + col]   = hC1[tt][m];
        }
      }
    }

    // out = tanh(h1 @ lin_w^T + lin_b); feed back as next decoder input
    v16h a0 = loadA(h1, 64, 0, lane);
    v16h a1 = loadA(h1, 64, 1, lane);
    v8f acc = splat8(blin[c]);
    acc = wmma16(a0, loadBlin(smem + OFF_LIN, 0, lane), acc);
    acc = wmma16(a1, loadBlin(smem + OFF_LIN, 1, lane), acc);
    if (c < 2){
#pragma unroll
      for (int m = 0; m < 8; ++m){
        float o = tanh_(acc[m]);
        xt[(mb + m) * 32 + c] = (_Float16)o;
        out[(size_t)(b0 + mb + m) * (TPRED * 2) + t * 2 + c] = o;
      }
    }
  }
}

extern "C" void kernel_launch(void* const* d_in, const int* in_sizes, int n_in,
                              void* d_out, int out_size, void* d_ws, size_t ws_size,
                              hipStream_t stream) {
  (void)in_sizes; (void)n_in; (void)d_ws; (void)ws_size; (void)out_size;
  const float* in_data = (const float*)d_in[0];
  const float* last    = (const float*)d_in[1];
  const float* e0_Wih = (const float*)d_in[2];
  const float* e0_Whh = (const float*)d_in[3];
  const float* e0_bih = (const float*)d_in[4];
  const float* e0_bhh = (const float*)d_in[5];
  const float* e1_Wih = (const float*)d_in[6];
  const float* e1_Whh = (const float*)d_in[7];
  const float* e1_bih = (const float*)d_in[8];
  const float* e1_bhh = (const float*)d_in[9];
  const float* d0_Wih = (const float*)d_in[10];
  const float* d0_Whh = (const float*)d_in[11];
  const float* d0_bih = (const float*)d_in[12];
  const float* d0_bhh = (const float*)d_in[13];
  const float* d1_Wih = (const float*)d_in[14];
  const float* d1_Whh = (const float*)d_in[15];
  const float* d1_bih = (const float*)d_in[16];
  const float* d1_bhh = (const float*)d_in[17];
  const float* lin_w  = (const float*)d_in[18];
  const float* lin_b  = (const float*)d_in[19];
  float* out = (float*)d_out;

  dim3 grid(BATCH / (WAVES * 16));   // 1024 blocks, 16 batch rows per wave
  seq2seq_gru_wmma<<<grid, NTHREADS, SMEM_BYTES, stream>>>(
      in_data, last,
      e0_Wih, e0_Whh, e0_bih, e0_bhh,
      e1_Wih, e1_Whh, e1_bih, e1_bhh,
      d0_Wih, d0_Whh, d0_bih, d0_bhh,
      d1_Wih, d1_Whh, d1_bih, d1_bhh,
      lin_w, lin_b, out);
}
